// GN_27006754358121
// MI455X (gfx1250) — matrix operand order repaired and checked by reading, not compiled, not run on hardware
//
#include <hip/hip_runtime.h>
#include <hip/hip_bf16.h>

typedef __attribute__((ext_vector_type(16))) __bf16   v16bf;
typedef __attribute__((ext_vector_type(8)))  float    v8f;
typedef __attribute__((ext_vector_type(4)))  float    f4raw;
typedef __attribute__((ext_vector_type(4)))  unsigned u4raw;

static __device__ __forceinline__ unsigned bfround(unsigned u) {
    return (u + 0x7fffu + ((u >> 16) & 1u)) >> 16;      // RNE to bf16 (high 16)
}
// pack two f32 -> two bf16 in one dword (low = a, high = b)
static __device__ __forceinline__ unsigned pk2(float a, float b) {
#if __has_builtin(__builtin_amdgcn_cvt_pk_bf16_f32)
    typedef __attribute__((ext_vector_type(2))) __bf16 v2bf;
    v2bf p = __builtin_amdgcn_cvt_pk_bf16_f32(a, b);
    return __builtin_bit_cast(unsigned, p);
#else
    union { float f; unsigned u; } x, y; x.f = a; y.f = b;
    return bfround(x.u) | (bfround(y.u) << 16);
#endif
}
static __device__ __forceinline__ __bf16 f2bf(float f) {
    union { float f; unsigned u; } v; v.f = f;
    unsigned short h = (unsigned short)bfround(v.u);
    return __builtin_bit_cast(__bf16, h);
}
// force global (AS1) vector load for gathered rows (ext_vector: no ctor issues)
typedef __attribute__((address_space(1))) const f4raw gf4_t;
static __device__ __forceinline__ f4raw gload4(const float* p) {
    return *(const gf4_t*)p;
}

// ---------------------------------------------------------------------------
// Weight pack: W[K,NB*16] f32 (row-major) -> fragment-ordered bf16 so that
//   frag(kc, nbi, lane) = dst + ((kc*NB + nbi)*32 + lane)*16  is one v16bf:
//   element e = W[(kc*32 + e + 16*(lane>>4))][nbi*16 + (lane&15)]
// ---------------------------------------------------------------------------
__global__ void k_packW(const float* __restrict__ src, __bf16* __restrict__ dst,
                        int K, int NB) {
    int g = blockIdx.x * 256 + threadIdx.x;
    int total = K * NB * 16;
    if (g >= total) return;
    int e    = g & 15;
    int t    = g >> 4;
    int lane = t & 31;
    int t2   = t >> 5;
    int nbi  = t2 % NB;
    int kc   = t2 / NB;
    int k    = kc * 32 + e + 16 * (lane >> 4);
    int col  = nbi * 16 + (lane & 15);
    dst[g] = f2bf(src[(size_t)k * (NB * 16) + col]);
}

// ---------------------------------------------------------------------------
// Generic fused MLP-GEMM: out[M,128] = relu( concat(segs)[M,K] @ W + bias ) (+resid)
// Segments are 128 wide; W fragment-packed (NB=8). Block 256 (8 waves),
// tile 64 x 128. One barrier pair per 128-wide segment; 16 WMMA per segment.
// ---------------------------------------------------------------------------
__global__ void __launch_bounds__(256)
fused_mlp(int M, int KSEG,
          const float* b0, const int* i0,
          const float* b1, const int* i1,
          const float* b2, const int* i2,
          const float* b3, const int* i3,
          const __bf16* __restrict__ W,      // fragment-packed [K/32][8][32][16]
          const float*  __restrict__ bias,   // [128]
          const float*  resid,               // nullptr or [M,128] (may alias out)
          float* __restrict__ out)           // [M,128]
{
    __shared__ __align__(16) __bf16 As[64][136];   // 64 rows x 128 K, 16B-aligned rows

    const int tid    = threadIdx.x;
    const int lane   = tid & 31;
    const int wave   = tid >> 5;
    const int hl     = lane >> 4;
    const int lanelo = lane & 15;
    const int mbase  = blockIdx.x * 64;

    // staging role: this thread owns row rr, k-run [q*8, q*8+8) of each 32-chunk
    const int rr = tid & 63;
    const int q  = tid >> 6;                 // 0..3
    const int m  = mbase + rr;

    const float* base[4] = { b0, b1, b2, b3 };
    const int*   idx [4] = { i0, i1, i2, i3 };

    v8f acc[4];
#pragma unroll
    for (int s = 0; s < 4; ++s)
#pragma unroll
        for (int j = 0; j < 8; ++j) acc[s][j] = 0.0f;

    const v16bf* Wp = (const v16bf*)W;

    for (int seg = 0; seg < KSEG; ++seg) {
        // resolve gather row for this thread's row, this segment
        const int* ii = idx[seg];
        int r = 0;
        if (m < M) r = ii ? ii[m] : m;
        const float* rowp = base[seg] + (size_t)r * 128 + q * 8;

        // stage the whole 64x128 segment tile (this thread: 8 f32 per chunk)
#pragma unroll
        for (int c = 0; c < 4; ++c) {
            f4raw v0 = gload4(rowp + c * 32);
            f4raw v1 = gload4(rowp + c * 32 + 4);
            u4raw pk;
            pk.x = pk2(v0.x, v0.y); pk.y = pk2(v0.z, v0.w);
            pk.z = pk2(v1.x, v1.y); pk.w = pk2(v1.z, v1.w);
            *(u4raw*)&As[rr][c * 32 + q * 8] = pk;
        }
        __syncthreads();

#pragma unroll
        for (int c = 0; c < 4; ++c) {
            const int kc = seg * 4 + c;
            v16bf bf = Wp[(size_t)(kc * 8 + wave) * 32 + lane];
#pragma unroll
            for (int s = 0; s < 4; ++s) {
                v16bf af;
                const __bf16* ap = &As[16 * s + lanelo][c * 32 + 16 * hl];
#pragma unroll
                for (int e = 0; e < 8; ++e) af[e] = ap[e];
#pragma unroll
                for (int e = 8; e < 16; ++e) af[e] = ap[e];
                acc[s] = __builtin_amdgcn_wmma_f32_16x16x32_bf16(
                             false, af, false, bf, (short)0, acc[s], false, false);
            }
        }
        __syncthreads();
    }

    const int   col = wave * 16 + lanelo;
    const float bv  = bias[col];
#pragma unroll
    for (int s = 0; s < 4; ++s) {
#pragma unroll
        for (int j = 0; j < 8; ++j) {
            int row = mbase + 16 * s + 8 * hl + j;
            if (row < M) {
                float v = fmaxf(acc[s][j] + bv, 0.0f);
                if (resid) v += resid[(size_t)row * 128 + col];
                out[(size_t)row * 128 + col] = v;
            }
        }
    }
}

// ---------------------------------------------------------------------------
// Fingerprint GEMM + where(node_type==0, gemm, rbf(feat3)) epilogue.
// x_fp[N,2048] @ fp_w[2048,64] -> x[:,0:64]. Block 128 (4 waves), tile 64x64,
// W fragment-packed with NB=4.  One barrier pair per 128-wide K segment.
// ---------------------------------------------------------------------------
__global__ void __launch_bounds__(128)
embed_gemm(const float* __restrict__ xfp, const __bf16* __restrict__ fpw,
           const float* __restrict__ feat3, const int* __restrict__ ntype,
           float* __restrict__ xout, int M)
{
    __shared__ __align__(16) __bf16 As[64][136];
    const int tid = threadIdx.x, lane = tid & 31, wave = tid >> 5;
    const int hl = lane >> 4, lanelo = lane & 15;
    const int mbase = blockIdx.x * 64;

    const int rr = tid & 63;
    const int q  = tid >> 6;                 // 0..1 : k-run [q*16, q*16+16) per chunk
    const int m  = mbase + rr;
    const int r  = (m < M) ? m : 0;
    const float* rowp = xfp + (size_t)r * 2048 + q * 16;

    v8f acc[4];
#pragma unroll
    for (int s = 0; s < 4; ++s)
#pragma unroll
        for (int j = 0; j < 8; ++j) acc[s][j] = 0.0f;

    const v16bf* Wp = (const v16bf*)fpw;

    for (int seg = 0; seg < 16; ++seg) {     // 16 segments of 128 K
#pragma unroll
        for (int c = 0; c < 4; ++c) {        // this thread: 16 f32 per chunk
            const float* p = rowp + (size_t)seg * 128 + c * 32;
            f4raw v0 = gload4(p);
            f4raw v1 = gload4(p + 4);
            f4raw v2 = gload4(p + 8);
            f4raw v3 = gload4(p + 12);
            u4raw pa, pb;
            pa.x = pk2(v0.x, v0.y); pa.y = pk2(v0.z, v0.w);
            pa.z = pk2(v1.x, v1.y); pa.w = pk2(v1.z, v1.w);
            pb.x = pk2(v2.x, v2.y); pb.y = pk2(v2.z, v2.w);
            pb.z = pk2(v3.x, v3.y); pb.w = pk2(v3.z, v3.w);
            *(u4raw*)&As[rr][c * 32 + q * 16]     = pa;
            *(u4raw*)&As[rr][c * 32 + q * 16 + 8] = pb;
        }
        __syncthreads();
#pragma unroll
        for (int c = 0; c < 4; ++c) {
            const int kc = seg * 4 + c;
            v16bf bf = Wp[(size_t)(kc * 4 + wave) * 32 + lane];
#pragma unroll
            for (int s = 0; s < 4; ++s) {
                v16bf af;
                const __bf16* ap = &As[16 * s + lanelo][c * 32 + 16 * hl];
#pragma unroll
                for (int e = 0; e < 8; ++e) af[e] = ap[e];
#pragma unroll
                for (int e = 8; e < 16; ++e) af[e] = ap[e];
                acc[s] = __builtin_amdgcn_wmma_f32_16x16x32_bf16(
                             false, af, false, bf, (short)0, acc[s], false, false);
            }
        }
        __syncthreads();
    }

    const int   col = wave * 16 + lanelo;        // 0..63
    const float cen = 0.15625f * (float)col;     // arange(0,10,10/64)
#pragma unroll
    for (int s = 0; s < 4; ++s) {
#pragma unroll
        for (int j = 0; j < 8; ++j) {
            int row = mbase + 16 * s + 8 * hl + j;
            if (row < M) {
                float d = feat3[row] - cen;
                float rb = __expf(-d * d * 0.04f);
                xout[(size_t)row * 128 + col] = (ntype[row] == 0) ? acc[s][j] : rb;
            }
        }
    }
}

// ---------------------------- elementwise helpers ---------------------------
__global__ void k_zero(float* p, int n) {
    int g = blockIdx.x * 256 + threadIdx.x; if (g < n) p[g] = 0.0f;
}
__global__ void k_gather_ea(const float* emb, const int* eidx, float* dst, int E) {
    int g = blockIdx.x * 256 + threadIdx.x;
    if (g < E * 128) { int e = g >> 7, c = g & 127; dst[g] = emb[eidx[e] * 128 + c]; }
}
__global__ void k_ebatch(const int* row, const int* batch, int* eb, int E) {
    int g = blockIdx.x * 256 + threadIdx.x; if (g < E) eb[g] = batch[row[g]];
}
__global__ void k_count(const int* idx, float* cnt, int n) {
    int g = blockIdx.x * 256 + threadIdx.x; if (g < n) atomicAdd(&cnt[idx[g]], 1.0f);
}
__global__ void k_invert(float* c, int n) {
    int g = blockIdx.x * 256 + threadIdx.x; if (g < n) c[g] = 1.0f / fmaxf(c[g], 1.0f);
}
__global__ void k_scatter(const float* src, const int* idx, float* dst, int rows) {
    int g = blockIdx.x * 256 + threadIdx.x;
    if (g < rows * 128) {
        int e = g >> 7, c = g & 127;
        atomicAdd(&dst[(size_t)idx[e] * 128 + c], src[g]);
    }
}
__global__ void k_scale_rows(float* d, const float* inv, int rows) {
    int g = blockIdx.x * 256 + threadIdx.x;
    if (g < rows * 128) d[g] *= inv[g >> 7];
}
__global__ void k_half2(const float* f4, float* x, int n) {
    int g = blockIdx.x * 256 + threadIdx.x;
    if (g < n * 64) {
        int nn = g >> 6, j = g & 63;
        float d = f4[nn] - 0.15625f * (float)j;
        x[(size_t)nn * 128 + 64 + j] = __expf(-d * d * 0.04f);
    }
}
__global__ void k_out(const float* __restrict__ x, const float* __restrict__ w,
                      const float* __restrict__ b, float* __restrict__ out, int M) {
    int node = (blockIdx.x * 256 + threadIdx.x) >> 5;
    int lane = threadIdx.x & 31;
    if (node >= M) return;
    float s = 0.0f;
#pragma unroll
    for (int t = 0; t < 4; ++t) {
        int k = lane + t * 32;
        s += x[(size_t)node * 128 + k] * w[k];
    }
#pragma unroll
    for (int off = 16; off > 0; off >>= 1) s += __shfl_down(s, off, 32);
    if (lane == 0) out[node] = s + b[0];
}

// ---------------------------------------------------------------------------
extern "C" void kernel_launch(void* const* d_in, const int* in_sizes, int n_in,
                              void* d_out, int out_size, void* d_ws, size_t ws_size,
                              hipStream_t stream)
{
    (void)n_in; (void)out_size; (void)ws_size;
    const float* x_fp  = (const float*)d_in[0];
    const float* feat3 = (const float*)d_in[1];
    const float* feat4 = (const float*)d_in[2];
    const int*   ntype = (const int*)d_in[3];
    const int*   eidx2 = (const int*)d_in[4];
    const int*   eaidx = (const int*)d_in[5];
    const int*   batch = (const int*)d_in[6];
    const float* eemb  = (const float*)d_in[7];
    const float* fpw   = (const float*)d_in[8];
    const float* ew1  = (const float*)d_in[9];   const float* eb1  = (const float*)d_in[10];
    const float* ew2  = (const float*)d_in[11];  const float* eb2  = (const float*)d_in[12];
    const float* n1w1 = (const float*)d_in[13];  const float* n1b1 = (const float*)d_in[14];
    const float* n1w2 = (const float*)d_in[15];  const float* n1b2 = (const float*)d_in[16];
    const float* n2w1 = (const float*)d_in[17];  const float* n2b1 = (const float*)d_in[18];
    const float* n2w2 = (const float*)d_in[19];  const float* n2b2 = (const float*)d_in[20];
    const float* gw1  = (const float*)d_in[21];  const float* gb1  = (const float*)d_in[22];
    const float* gw2  = (const float*)d_in[23];  const float* gb2  = (const float*)d_in[24];
    const float* outw = (const float*)d_in[25];  const float* outb = (const float*)d_in[26];

    const int N = in_sizes[1];      // feat3
    const int E = in_sizes[5];      // edge_attr_idx
    const int B = 64, L = 3;
    const int* erow = eidx2;
    const int* ecol = eidx2 + E;

    // ---- workspace carving ----
    size_t off = 0;
    auto carve = [&](size_t bytes) -> char* {
        char* p = (char*)d_ws + off;
        off += (bytes + 255) & ~(size_t)255;
        return p;
    };
    float* E0   = (float*)carve((size_t)E * 128 * 4);
    float* E1   = (float*)carve((size_t)E * 128 * 4);
    float* X0   = (float*)carve((size_t)N * 128 * 4);
    float* X1   = (float*)carve((size_t)N * 128 * 4);
    float* AGG  = (float*)carve((size_t)N * 128 * 4);
    float* U0   = (float*)carve((size_t)B * 128 * 4);
    float* U1   = (float*)carve((size_t)B * 128 * 4);
    float* SXB  = (float*)carve((size_t)B * 128 * 4);
    float* INVC = (float*)carve((size_t)N * 4);
    float* INVB = (float*)carve((size_t)B * 4);
    int*   EB   = (int*)carve((size_t)E * 4);
    __bf16* WFP  = (__bf16*)carve((size_t)2048 * 64 * 2);
    __bf16* WE1  = (__bf16*)carve((size_t)L * 512 * 128 * 2);
    __bf16* WE2  = (__bf16*)carve((size_t)L * 128 * 128 * 2);
    __bf16* WN1a = (__bf16*)carve((size_t)L * 256 * 128 * 2);
    __bf16* WN1b = (__bf16*)carve((size_t)L * 128 * 128 * 2);
    __bf16* WN2a = (__bf16*)carve((size_t)L * 384 * 128 * 2);
    __bf16* WN2b = (__bf16*)carve((size_t)L * 128 * 128 * 2);
    __bf16* WGa  = (__bf16*)carve((size_t)L * 256 * 128 * 2);
    __bf16* WGb  = (__bf16*)carve((size_t)L * 128 * 128 * 2);

    auto cdiv = [](int a, int b) { return (a + b - 1) / b; };
    auto pack = [&](const float* s, __bf16* d, int K, int layer) {
        const float* sp = s + (size_t)layer * K * 128;
        __bf16*      dp = d + (size_t)layer * K * 128;
        k_packW<<<cdiv(K * 128, 256), 256, 0, stream>>>(sp, dp, K, 8);
    };

    // ---- one-time prep ----
    k_packW<<<cdiv(2048 * 64, 256), 256, 0, stream>>>(fpw, WFP, 2048, 4);
    for (int i = 0; i < L; ++i) {
        pack(ew1,  WE1,  512, i);
        pack(ew2,  WE2,  128, i);
        pack(n1w1, WN1a, 256, i);
        pack(n1w2, WN1b, 128, i);
        pack(n2w1, WN2a, 384, i);
        pack(n2w2, WN2b, 128, i);
        pack(gw1,  WGa,  256, i);
        pack(gw2,  WGb,  128, i);
    }

    k_zero<<<cdiv(N, 256), 256, 0, stream>>>(INVC, N);
    k_zero<<<cdiv(B, 256), 256, 0, stream>>>(INVB, B);
    k_count<<<cdiv(E, 256), 256, 0, stream>>>(ecol, INVC, E);
    k_count<<<cdiv(N, 256), 256, 0, stream>>>(batch, INVB, N);
    k_invert<<<cdiv(N, 256), 256, 0, stream>>>(INVC, N);
    k_invert<<<cdiv(B, 256), 256, 0, stream>>>(INVB, B);
    k_ebatch<<<cdiv(E, 256), 256, 0, stream>>>(erow, batch, EB, E);
    k_gather_ea<<<cdiv(E * 128, 256), 256, 0, stream>>>(eemb, eaidx, E0, E);
    k_zero<<<cdiv(B * 128, 256), 256, 0, stream>>>(U0, B * 128);

    embed_gemm<<<cdiv(N, 64), 128, 0, stream>>>(x_fp, WFP, feat3, ntype, X0, N);
    k_half2<<<cdiv(N * 64, 256), 256, 0, stream>>>(feat4, X0, N);

    // ---- 3 GraphNet layers ----
    float* xcur = X0; float* xoth = X1;
    float* ucur = U0; float* uoth = U1;
    float* ea   = E0; float* hbuf = E1;
    const int gE = cdiv(E, 64), gN = cdiv(N, 64), gB = cdiv(B, 64);
    const float* NF = nullptr; const int* NI = nullptr;

    for (int i = 0; i < L; ++i) {
        const __bf16* we1  = WE1  + (size_t)i * 512 * 128;
        const __bf16* we2  = WE2  + (size_t)i * 128 * 128;
        const __bf16* wn1a = WN1a + (size_t)i * 256 * 128;
        const __bf16* wn1b = WN1b + (size_t)i * 128 * 128;
        const __bf16* wn2a = WN2a + (size_t)i * 384 * 128;
        const __bf16* wn2b = WN2b + (size_t)i * 128 * 128;
        const __bf16* wga  = WGa  + (size_t)i * 256 * 128;
        const __bf16* wgb  = WGb  + (size_t)i * 128 * 128;

        // EdgeModel: h = relu([x[row],x[col],ea,u[ebatch]] @ we1 + b1); ea = h + relu(h@we2+b2)
        fused_mlp<<<gE, 256, 0, stream>>>(E, 4, xcur, erow, xcur, ecol, ea, NI, ucur, EB,
                                          we1, eb1 + i * 128, NF, hbuf);
        fused_mlp<<<gE, 256, 0, stream>>>(E, 1, hbuf, NI, NF, NI, NF, NI, NF, NI,
                                          we2, eb2 + i * 128, hbuf, hbuf);
        { float* t = ea; ea = hbuf; hbuf = t; }   // new edge_attr in ea

        // NodeModel message: m = mlp([x[row], ea])
        fused_mlp<<<gE, 256, 0, stream>>>(E, 2, xcur, erow, ea, NI, NF, NI, NF, NI,
                                          wn1a, n1b1 + i * 128, NF, hbuf);
        fused_mlp<<<gE, 256, 0, stream>>>(E, 1, hbuf, NI, NF, NI, NF, NI, NF, NI,
                                          wn1b, n1b2 + i * 128, hbuf, hbuf);

        // agg = scatter_mean(m, col, N)
        k_zero<<<cdiv(N * 128, 256), 256, 0, stream>>>(AGG, N * 128);
        k_scatter<<<cdiv(E * 128, 256), 256, 0, stream>>>(hbuf, ecol, AGG, E);
        k_scale_rows<<<cdiv(N * 128, 256), 256, 0, stream>>>(AGG, INVC, N);

        // Node update: x = mlp([x, agg, u[batch]])
        fused_mlp<<<gN, 256, 0, stream>>>(N, 3, xcur, NI, AGG, NI, ucur, batch, NF, NI,
                                          wn2a, n2b1 + i * 128, NF, xoth);
        fused_mlp<<<gN, 256, 0, stream>>>(N, 1, xoth, NI, NF, NI, NF, NI, NF, NI,
                                          wn2b, n2b2 + i * 128, xoth, xoth);
        { float* t = xcur; xcur = xoth; xoth = t; }

        // GlobalModel: u = mlp([u, scatter_mean(x, batch, B)])
        k_zero<<<cdiv(B * 128, 256), 256, 0, stream>>>(SXB, B * 128);
        k_scatter<<<cdiv(N * 128, 256), 256, 0, stream>>>(xcur, batch, SXB, N);
        k_scale_rows<<<cdiv(B * 128, 256), 256, 0, stream>>>(SXB, INVB, B);
        fused_mlp<<<gB, 256, 0, stream>>>(B, 2, ucur, NI, SXB, NI, NF, NI, NF, NI,
                                          wga, gb1 + i * 128, NF, uoth);
        fused_mlp<<<gB, 256, 0, stream>>>(B, 1, uoth, NI, NF, NI, NF, NI, NF, NI,
                                          wgb, gb2 + i * 128, uoth, uoth);
        { float* t = ucur; ucur = uoth; uoth = t; }
    }

    // out = x @ out_w + out_b
    k_out<<<cdiv(N * 32, 256), 256, 0, stream>>>(xcur, outw, outb, (float*)d_out, N);
}
